// Eagle3Attention_60026462929412
// MI455X (gfx1250) — matrix operand order repaired and checked by reading, not compiled, not run on hardware
//
#include <hip/hip_runtime.h>
#include <hip/hip_bf16.h>
#include <stdint.h>

#define S_LEN 2048
#define IN_SZ 5760        // 2*HIDDEN
#define HID   2880
#define NH    64
#define NKV   8
#define HD    64
#define QDIM  (NH * HD)   // 4096
#define KVDIM (NKV * HD)  // 512
#define P_STR 72          // LDS P-tile row stride (bf16 elems); 144B rows, 16B aligned

typedef __bf16 bf16;
typedef __attribute__((ext_vector_type(16))) __bf16 v16bf;
typedef __attribute__((ext_vector_type(8)))  __bf16 v8bf;
typedef __attribute__((ext_vector_type(8)))  float  v8f;

static __device__ inline v8f wmma_bf16(v16bf a, v16bf b, v8f c) {
  // D = A(16x32) * B(32x16) + C, f32 accumulate
  return __builtin_amdgcn_wmma_f32_16x16x32_bf16(false, a, false, b, (short)0, c,
                                                 false, false);
}

// ---- DPP16 16-lane row reductions (pure VALU, no LDS traffic) ----
template <int CTRL>
static __device__ inline float dpp_mov(float x) {
  int i = __builtin_bit_cast(int, x);
  int r = __builtin_amdgcn_update_dpp(0, i, CTRL, 0xF, 0xF, true);
  return __builtin_bit_cast(float, r);
}
static __device__ inline float row_max16(float x) {
  x = fmaxf(x, dpp_mov<0xB1>(x));   // quad_perm [1,0,3,2]  (xor 1)
  x = fmaxf(x, dpp_mov<0x4E>(x));   // quad_perm [2,3,0,1]  (xor 2)
  x = fmaxf(x, dpp_mov<0x124>(x));  // row_ror:4
  x = fmaxf(x, dpp_mov<0x128>(x));  // row_ror:8
  return x;
}
static __device__ inline float row_sum16(float x) {
  x += dpp_mov<0xB1>(x);
  x += dpp_mov<0x4E>(x);
  x += dpp_mov<0x124>(x);
  x += dpp_mov<0x128>(x);
  return x;
}

// A-operand fragment (16 rows x 32 K), source row-major [row][K], leading dim ld.
// lane = half*16 + r holds row r: elems 0..7 = K(half*8..), elems 8..15 = K(16+half*8..)
template <typename T>
static __device__ inline v16bf load_fragA(const T* p0, int ld, int lane) {
  int r = lane & 15, hf = lane >> 4;
  const T* p = p0 + (size_t)r * ld + hf * 8;
  union { v16bf v; v8bf q[2]; } u;
  u.q[0] = *(const v8bf*)(p);
  u.q[1] = *(const v8bf*)(p + 16);
  return u.v;
}

// B-operand fragment (32 K x 16 cols), source stored as N x K row-major ("Bt").
// lane = half*16 + r holds column r: elems 0..7 = K(half*8..), elems 8..15 = K(16+half*8..)
template <typename T>
static __device__ inline v16bf load_fragB(const T* p0, int ld, int lane) {
  int r = lane & 15, hf = lane >> 4;
  const T* p = p0 + (size_t)r * ld + hf * 8;
  union { v16bf v; v8bf q[2]; } u;
  u.q[0] = *(const v8bf*)(p);
  u.q[1] = *(const v8bf*)(p + 16);
  return u.v;
}

// ---------------------------------------------------------------- converters
__global__ void k_f2bf(const float* __restrict__ s, bf16* __restrict__ d, size_t n) {
  size_t i = (size_t)blockIdx.x * blockDim.x + threadIdx.x;
  size_t stride = (size_t)gridDim.x * blockDim.x;
  for (; i < n; i += stride) d[i] = (bf16)s[i];
}

// src[R][C] (f32) -> dst[C][R] (bf16), tiled transpose
__global__ __launch_bounds__(256) void k_tr_f2bf(const float* __restrict__ src,
                                                 bf16* __restrict__ dst,
                                                 int R, int C) {
  __shared__ float tile[32][33];
  int c0 = blockIdx.x * 32, r0 = blockIdx.y * 32;
  int tx = threadIdx.x, ty = threadIdx.y;
#pragma unroll
  for (int yy = ty; yy < 32; yy += 8) {
    int rr = r0 + yy, cc = c0 + tx;
    tile[yy][tx] = (rr < R && cc < C) ? src[(size_t)rr * C + cc] : 0.0f;
  }
  __syncthreads();
#pragma unroll
  for (int yy = ty; yy < 32; yy += 8) {
    int cc = c0 + yy, rr = r0 + tx;
    if (cc < C && rr < R) dst[(size_t)cc * R + rr] = (bf16)tile[tx][yy];
  }
}

// src[R][C] (bf16) -> dst[C][R] (bf16)
__global__ __launch_bounds__(256) void k_tr_bf(const bf16* __restrict__ src,
                                               bf16* __restrict__ dst,
                                               int R, int C) {
  __shared__ float tile[32][33];
  int c0 = blockIdx.x * 32, r0 = blockIdx.y * 32;
  int tx = threadIdx.x, ty = threadIdx.y;
#pragma unroll
  for (int yy = ty; yy < 32; yy += 8) {
    int rr = r0 + yy, cc = c0 + tx;
    tile[yy][tx] = (rr < R && cc < C) ? (float)src[(size_t)rr * C + cc] : 0.0f;
  }
  __syncthreads();
#pragma unroll
  for (int yy = ty; yy < 32; yy += 8) {
    int cc = c0 + yy, rr = r0 + tx;
    if (cc < C && rr < R) dst[(size_t)cc * R + rr] = (bf16)tile[tx][yy];
  }
}

// ---------------------------------------------------------------------- GEMM
// C[M][N] = A[M][K] * Bt[N][K]^T ; one wave -> 64x64 tile (4x4 WMMA tiles),
// 4 waves arranged 2x2 -> block tile 128x128 (balanced A/B reuse in cache).
template <bool F32OUT>
__global__ __launch_bounds__(128) void k_gemm(const bf16* __restrict__ A,
                                              const bf16* __restrict__ Bt,
                                              void* __restrict__ Cv,
                                              int M, int N, int K) {
  int lane = threadIdx.x & 31;
  int wave = threadIdx.x >> 5;
  int row0 = blockIdx.y * 128 + (wave >> 1) * 64;
  int col0 = (blockIdx.x * 2 + (wave & 1)) * 64;
  if (col0 >= N || row0 >= M) return;

  v8f acc[4][4] = {};
  for (int k0 = 0; k0 < K; k0 += 32) {
    v16bf a[4], b[4];
#pragma unroll
    for (int i = 0; i < 4; ++i)
      a[i] = load_fragA(A + (size_t)(row0 + 16 * i) * K + k0, K, lane);
#pragma unroll
    for (int j = 0; j < 4; ++j)
      b[j] = load_fragB(Bt + (size_t)(col0 + 16 * j) * K + k0, K, lane);
#pragma unroll
    for (int i = 0; i < 4; ++i)
#pragma unroll
      for (int j = 0; j < 4; ++j)
        acc[i][j] = wmma_bf16(a[i], b[j], acc[i][j]);
  }

  int r = lane & 15, hf = lane >> 4;
#pragma unroll
  for (int i = 0; i < 4; ++i) {
#pragma unroll
    for (int j = 0; j < 4; ++j) {
#pragma unroll
      for (int e = 0; e < 8; ++e) {
        int rr = row0 + 16 * i + e + 8 * hf;
        int cc = col0 + 16 * j + r;
        if constexpr (F32OUT) {
          ((float*)Cv)[(size_t)rr * N + cc] = acc[i][j][e];
        } else {
          ((bf16*)Cv)[(size_t)rr * N + cc] = (bf16)acc[i][j][e];
        }
      }
    }
  }
}

// ---------------------------------------------------------------------- RoPE
// x[S][nheads*64] in-place; each thread handles the (d, d+32) pair.
__global__ void k_rope(bf16* __restrict__ x, const float* __restrict__ cosb,
                       const float* __restrict__ sinb, int nheads, float scale,
                       size_t total) {
  size_t idx = (size_t)blockIdx.x * blockDim.x + threadIdx.x;
  if (idx >= total) return;
  int d2 = (int)(idx & 31);
  size_t t = idx >> 5;
  int head = (int)(t % nheads);
  int s = (int)(t / nheads);
  size_t base = ((size_t)s * nheads + head) * 64;
  float c1 = cosb[s * 64 + d2],      s1 = sinb[s * 64 + d2];
  float c2 = cosb[s * 64 + d2 + 32], s2 = sinb[s * 64 + d2 + 32];
  float x1 = (float)x[base + d2];
  float x2 = (float)x[base + d2 + 32];
  x[base + d2]      = (bf16)((x1 * c1 - x2 * s1) * scale);
  x[base + d2 + 32] = (bf16)((x2 * c2 + x1 * s2) * scale);
}

// ------------------------------------------------------------ flash attention
// grid (S/64, NH), 4 waves/block; wave = 16 query rows, 64 keys per step.
__global__ __launch_bounds__(128) void k_flash(const bf16* __restrict__ Q,
                                               const bf16* __restrict__ Kb,
                                               const bf16* __restrict__ Vt,
                                               bf16* __restrict__ AO) {
  __shared__ __attribute__((aligned(16))) bf16 plds[4][16][P_STR];
  int lane = threadIdx.x & 31;
  int wave = threadIdx.x >> 5;
  int r = lane & 15, hf = lane >> 4;
  int h = blockIdx.y;
  int kvh = h >> 3;  // GQA: 8 query heads per kv head
  int qbase = blockIdx.x * 64 + wave * 16;

  const bf16* qp = Q + (size_t)qbase * QDIM + h * HD;
  v16bf aq0 = load_fragA(qp, QDIM, lane);       // d 0..31   (q pre-scaled by 1/8)
  v16bf aq1 = load_fragA(qp + 32, QDIM, lane);  // d 32..63

  v8f acc[4] = {};
  float m[8], l[8];
#pragma unroll
  for (int i = 0; i < 8; ++i) { m[i] = -1e30f; l[i] = 0.0f; }

  const bf16* kp = Kb + kvh * HD;                     // [S][KVDIM]
  const bf16* vp = Vt + (size_t)(kvh * HD) * S_LEN;   // [KVDIM][S]

  for (int kb = 0; kb < S_LEN; kb += 64) {
    // ---- scores: 4 tiles of 16 keys, contraction over d (2 WMMAs each)
    v8f st[4] = {};
#pragma unroll
    for (int t = 0; t < 4; ++t) {
      v16bf b0 = load_fragB(kp + (size_t)(kb + 16 * t) * KVDIM, KVDIM, lane);
      v16bf b1 = load_fragB(kp + (size_t)(kb + 16 * t) * KVDIM + 32, KVDIM, lane);
      st[t] = wmma_bf16(aq0, b0, st[t]);
      st[t] = wmma_bf16(aq1, b1, st[t]);
    }

    // ---- online softmax (row i+8*hf per C-layout); DPP row reductions
#pragma unroll
    for (int i = 0; i < 8; ++i) {
      float mx = fmaxf(fmaxf(st[0][i], st[1][i]), fmaxf(st[2][i], st[3][i]));
      mx = row_max16(mx);
      float mn = fmaxf(m[i], mx);
      float corr = __expf(m[i] - mn);
      float p0 = __expf(st[0][i] - mn);
      float p1 = __expf(st[1][i] - mn);
      float p2 = __expf(st[2][i] - mn);
      float p3 = __expf(st[3][i] - mn);
      float rs = row_sum16((p0 + p1) + (p2 + p3));
      l[i] = l[i] * corr + rs;
      m[i] = mn;
      acc[0][i] *= corr; acc[1][i] *= corr; acc[2][i] *= corr; acc[3][i] *= corr;
      bf16* row = &plds[wave][i + 8 * hf][0];
      row[r]      = (bf16)p0;   // C-layout -> row-major LDS (16x64 P tile)
      row[16 + r] = (bf16)p1;
      row[32 + r] = (bf16)p2;
      row[48 + r] = (bf16)p3;
    }

    // ---- P (16x64) back as two A-fragments; PV with Vt as B (N=d, K=key)
    v16bf ap0 = load_fragA(&plds[wave][0][0], P_STR, lane);       // keys kb..kb+31
    v16bf ap1 = load_fragA(&plds[wave][0][0] + 32, P_STR, lane);  // keys kb+32..63
#pragma unroll
    for (int dt = 0; dt < 4; ++dt) {
      v16bf bv0 = load_fragB(vp + (size_t)(16 * dt) * S_LEN + kb, S_LEN, lane);
      v16bf bv1 = load_fragB(vp + (size_t)(16 * dt) * S_LEN + kb + 32, S_LEN, lane);
      acc[dt] = wmma_bf16(ap0, bv0, acc[dt]);
      acc[dt] = wmma_bf16(ap1, bv1, acc[dt]);
    }
  }

#pragma unroll
  for (int i = 0; i < 8; ++i) {
    int row = qbase + i + 8 * hf;
    float inv = 1.0f / l[i];
    bf16* o = AO + (size_t)row * QDIM + h * HD;
    o[r]      = (bf16)(acc[0][i] * inv);
    o[16 + r] = (bf16)(acc[1][i] * inv);
    o[32 + r] = (bf16)(acc[2][i] * inv);
    o[48 + r] = (bf16)(acc[3][i] * inv);
  }
}

// -------------------------------------------------------------------- launch
extern "C" void kernel_launch(void* const* d_in, const int* in_sizes, int n_in,
                              void* d_out, int out_size, void* d_ws, size_t ws_size,
                              hipStream_t stream) {
  const float* hs   = (const float*)d_in[0];
  const float* cosb = (const float*)d_in[1];
  const float* sinb = (const float*)d_in[2];
  const float* wq   = (const float*)d_in[3];
  const float* wk   = (const float*)d_in[4];
  const float* wv   = (const float*)d_in[5];
  const float* wo   = (const float*)d_in[6];
  (void)in_sizes; (void)n_in; (void)out_size; (void)ws_size;

  char* ws = (char*)d_ws;
  size_t off = 0;
  auto alloc = [&](size_t bytes) {
    char* p = ws + off;
    off += (bytes + 255) & ~(size_t)255;
    return p;
  };
  bf16* Xbf = (bf16*)alloc((size_t)S_LEN * IN_SZ * 2);
  bf16* WqT = (bf16*)alloc((size_t)QDIM  * IN_SZ * 2);
  bf16* WkT = (bf16*)alloc((size_t)KVDIM * IN_SZ * 2);
  bf16* WvT = (bf16*)alloc((size_t)KVDIM * IN_SZ * 2);
  bf16* WoT = (bf16*)alloc((size_t)HID   * QDIM  * 2);
  bf16* Qb  = (bf16*)alloc((size_t)S_LEN * QDIM  * 2);
  bf16* Kb  = (bf16*)alloc((size_t)S_LEN * KVDIM * 2);
  bf16* Vb  = (bf16*)alloc((size_t)S_LEN * KVDIM * 2);
  bf16* Vt  = (bf16*)alloc((size_t)KVDIM * S_LEN * 2);
  bf16* AO  = (bf16*)alloc((size_t)S_LEN * QDIM  * 2);

  // fp32 -> bf16 (weights pre-transposed to N x K for WMMA B-operand loads)
  {
    size_t n = (size_t)S_LEN * IN_SZ;
    k_f2bf<<<(int)((n + 1023) / 1024), 1024, 0, stream>>>(hs, Xbf, n);
  }
  dim3 tb(32, 8);
  k_tr_f2bf<<<dim3(QDIM / 32,  IN_SZ / 32), tb, 0, stream>>>(wq, WqT, IN_SZ, QDIM);
  k_tr_f2bf<<<dim3(KVDIM / 32, IN_SZ / 32), tb, 0, stream>>>(wk, WkT, IN_SZ, KVDIM);
  k_tr_f2bf<<<dim3(KVDIM / 32, IN_SZ / 32), tb, 0, stream>>>(wv, WvT, IN_SZ, KVDIM);
  k_tr_f2bf<<<dim3(HID / 32,   QDIM / 32),  tb, 0, stream>>>(wo, WoT, QDIM, HID);

  // QKV projections (bf16 WMMA, f32 accumulate); block tile 128x128
  k_gemm<false><<<dim3(QDIM / 128,  S_LEN / 128), 128, 0, stream>>>(Xbf, WqT, Qb, S_LEN, QDIM,  IN_SZ);
  k_gemm<false><<<dim3(KVDIM / 128, S_LEN / 128), 128, 0, stream>>>(Xbf, WkT, Kb, S_LEN, KVDIM, IN_SZ);
  k_gemm<false><<<dim3(KVDIM / 128, S_LEN / 128), 128, 0, stream>>>(Xbf, WvT, Vb, S_LEN, KVDIM, IN_SZ);

  // RoPE (softmax scale 1/sqrt(64) folded into q)
  {
    size_t nq = (size_t)S_LEN * NH * 32;
    k_rope<<<(int)((nq + 255) / 256), 256, 0, stream>>>(Qb, cosb, sinb, NH, 0.125f, nq);
    size_t nk = (size_t)S_LEN * NKV * 32;
    k_rope<<<(int)((nk + 255) / 256), 256, 0, stream>>>(Kb, cosb, sinb, NKV, 1.0f, nk);
  }

  // V -> Vt [KVDIM][S] so PV B-operand loads are contiguous
  k_tr_bf<<<dim3(KVDIM / 32, S_LEN / 32), tb, 0, stream>>>(Vb, Vt, S_LEN, KVDIM);

  // flash attention -> AO [S][QDIM] bf16
  k_flash<<<dim3(S_LEN / 64, NH), 128, 0, stream>>>(Qb, Kb, Vt, AO);

  // output projection -> f32 d_out [S][HID]
  k_gemm<true><<<dim3((HID + 127) / 128, S_LEN / 128), 128, 0, stream>>>(AO, WoT, d_out, S_LEN, HID, QDIM);
}
